// LoRAConv2d_86380382257238
// MI455X (gfx1250) — matrix operand order repaired and checked by reading, not compiled, not run on hardware
//
#include <hip/hip_runtime.h>
#include <hip/hip_bf16.h>

typedef __attribute__((ext_vector_type(16))) _Float16 v16h;
typedef __attribute__((ext_vector_type(8)))  _Float16 v8h;
typedef __attribute__((ext_vector_type(8)))  float    v8f;

#define CIN   256
#define COUT  256
#define HW    64
#define NIMG  16
#define RANK  16
#define SCALING 2.0f          // alpha/rank = 32/16
#define SPATIAL (HW*HW)       // 4096
#define MTOT  (NIMG*SPATIAL)  // 65536
#define BM    128
#define BN    128
#define BK    64              // K-chunk: 64 channels
#define NKC   36              // 9 taps x 4 channel-chunks of 64
#define APAD  72              // halves per A row  (64 + 8 pad) -> 144B stride
#define BPAD  72              // halves per B col  (64 + 8 pad) -> 144B stride
#define CPAD  129             // floats per C row in LDS epilogue

// ---------------------------------------------------------------------------
// Kernel 1: fused weight  W' = conv_w + SCALING*(B@A), cast f16, laid out as
//           Wt[tap][chunk][oc][klocal(64)]  so a B-tile is a contiguous copy.
// ---------------------------------------------------------------------------
__global__ void fuse_weights_kernel(const float* __restrict__ conv_w,
                                    const float* __restrict__ lora_A,
                                    const float* __restrict__ lora_B,
                                    _Float16* __restrict__ Wt) {
    int idx = blockIdx.x * 256 + threadIdx.x;      // 9*4*256*64 = 589824
    int klocal = idx & 63;
    int oc     = (idx >> 6) & 255;
    int chunk  = (idx >> 14) & 3;
    int tap    = idx >> 16;                        // 0..8
    int c = chunk * 64 + klocal;

    float w = conv_w[oc * (CIN*9) + c * 9 + tap];  // OIHW flat, tap = kh*3+kw
    float dot = 0.f;
#pragma unroll
    for (int r = 0; r < RANK; ++r)
        dot += lora_B[oc * RANK + r] * lora_A[r * (CIN*9) + c * 9 + tap];
    Wt[idx] = (_Float16)(w + SCALING * dot);
}

// ---------------------------------------------------------------------------
// Kernel 2: transpose+convert x (NCHW f32) -> xh[n][s][c] f16 (channels
// innermost) via LDS tile, so im2col A-tiles become contiguous runs.
// ---------------------------------------------------------------------------
__global__ void convert_x_kernel(const float* __restrict__ x,
                                 _Float16* __restrict__ xh) {
    __shared__ _Float16 L[64 * 80];                 // [s][c] tile, padded rows
    const int t  = threadIdx.x;
    const int b  = blockIdx.x;                      // 16n * 64st * 4ct = 4096
    const int ct = b & 3, st = (b >> 2) & 63, n = b >> 8;
    const int s0 = st << 6, c0 = ct << 6;

    const int si = t & 63, cb = t >> 6;
#pragma unroll
    for (int i = 0; i < 16; ++i) {
        int ci = cb + (i << 2);
        float v = x[(((n << 8) + c0 + ci) << 12) + s0 + si];  // coalesced in s
        L[si * 80 + ci] = (_Float16)v;
    }
    __syncthreads();
    const int sj = t >> 2, cj = (t & 3) << 4;
#pragma unroll
    for (int j = 0; j < 2; ++j) {
        v8h d = *(const v8h*)&L[sj * 80 + cj + j * 8];
        *(v8h*)&xh[(((n << 12) + s0 + sj) << 8) + c0 + cj + j * 8] = d;
    }
}

// ---------------------------------------------------------------------------
// Kernel 3: implicit-GEMM conv, V_WMMA_F32_16X16X32_F16, double-buffered LDS
// fed by GLOBAL_LOAD_ASYNC_TO_LDS_B128 (ASYNCcnt pipeline), K-chunk = 64.
// ---------------------------------------------------------------------------
struct SharedMem {
    union {
        struct {
            _Float16 A[2][BM * APAD];   // [row][k], 144B row stride
            _Float16 B[2][BN * BPAD];   // [col][k], 144B col stride
        } ab;
        float C[BM * CPAD];             // epilogue staging
    } u;
};

__device__ __forceinline__ void async_b128(unsigned lds_off, const void* g) {
    asm volatile("global_load_async_to_lds_b128 %0, %1, off"
                 :: "v"(lds_off), "v"(g) : "memory");
}

static __device__ __forceinline__ v16h make_frag(const _Float16* p0, const _Float16* p1) {
    v8h lo = *(const v8h*)p0;
    v8h hi = *(const v8h*)p1;
    return __builtin_shufflevector(lo, hi, 0,1,2,3,4,5,6,7,8,9,10,11,12,13,14,15);
}

__global__ void __launch_bounds__(256, 2)
lora_conv_wmma_kernel(const _Float16* __restrict__ xh,
                      const _Float16* __restrict__ Wt,
                      const float* __restrict__ bias,
                      float* __restrict__ out) {
    __shared__ SharedMem smem;

    const int tid     = threadIdx.x;
    const int lane    = tid & 31;
    const int wave    = tid >> 5;
    const int wave_m  = wave & 3;          // 4 M-groups of 32 rows
    const int wave_n  = wave >> 2;         // 2 N-groups of 64 cols
    const int block_m = blockIdx.x * BM;
    const int nb      = blockIdx.y * BN;
    const int l15 = lane & 15;
    const int lhi = lane >> 4;

    // ---- per-thread staging constants (2 threads per row/col, 64B each) ----
    const int arow  = tid >> 1;
    const int ahalf = tid & 1;             // 32 halves (64B) each
    const int am    = block_m + arow;
    const int an    = am >> 12;
    const int as_   = am & 4095;
    const int aoh   = as_ >> 6, aow = as_ & 63;
    const int bcol  = tid >> 1;
    const int bkh   = (tid & 1) << 5;      // 0 or 32 halves
    const _Float16* gb_base = Wt + (((long)nb + bcol) << 6) + bkh;

    auto issue = [&](int kc) {
        _Float16* Ab = smem.u.ab.A[kc & 1];
        _Float16* Bb = smem.u.ab.B[kc & 1];
        const int tap = kc >> 2, chunk = kc & 3;
        const int q  = (tap * 11) >> 5;          // tap/3
        const int ih = aoh + q - 1;
        const int iw = aow + (tap - 3 * q) - 1;
        _Float16* ldsA = &Ab[arow * APAD + (ahalf << 5)];
        if ((unsigned)ih < HW && (unsigned)iw < HW) {
            const _Float16* g = xh + (an << 20) + (((ih << 6) + iw) << 8)
                              + (chunk << 6) + (ahalf << 5);
            unsigned ao = (unsigned)(uintptr_t)ldsA;
            async_b128(ao,       g);
            async_b128(ao + 16u, g + 8);
            async_b128(ao + 32u, g + 16);
            async_b128(ao + 48u, g + 24);
        } else {
            v8h z = {};
            *(v8h*)(ldsA)      = z;
            *(v8h*)(ldsA + 8)  = z;
            *(v8h*)(ldsA + 16) = z;
            *(v8h*)(ldsA + 24) = z;
        }
        const _Float16* g2 = gb_base + ((long)kc << 14);   // +COUT*64 halves/kc
        _Float16* ldsB = &Bb[bcol * BPAD + bkh];
        unsigned bo = (unsigned)(uintptr_t)ldsB;
        async_b128(bo,       g2);
        async_b128(bo + 16u, g2 + 8);
        async_b128(bo + 32u, g2 + 16);
        async_b128(bo + 48u, g2 + 24);
    };

    v8f acc[8];
#pragma unroll
    for (int i = 0; i < 8; ++i) acc[i] = (v8f){0,0,0,0,0,0,0,0};

    issue(0);                               // prologue fill of buffer 0
    for (int kc = 0; kc < NKC; ++kc) {
        asm volatile("s_wait_asynccnt 0x0" ::: "memory");  // buf[kc&1] landed
        __syncthreads();                    // all waves done with buf[(kc+1)&1]
        if (kc + 1 < NKC) issue(kc + 1);    // prefetch into other buffer

        const _Float16* Ab = smem.u.ab.A[kc & 1];
        const _Float16* Bb = smem.u.ab.B[kc & 1];

#pragma unroll
        for (int s = 0; s < 2; ++s) {       // two K=32 steps within the chunk
            v16h afrag[2];
#pragma unroll
            for (int mi = 0; mi < 2; ++mi) {
                int aro = wave_m * 32 + mi * 16 + l15;
                int kb  = (s << 5) + (lhi << 3);         // s*32 + 0/8
                const _Float16* pa = &Ab[aro * APAD + kb];
                afrag[mi] = make_frag(pa, pa + 16);
            }
            v16h bfrag[4];
#pragma unroll
            for (int ni = 0; ni < 4; ++ni) {
                int bco = wave_n * 64 + ni * 16 + l15;
                int ks  = (s << 5) + (lhi << 3);         // s*32 + 0/8
                const _Float16* pb = &Bb[bco * BPAD + ks];
                bfrag[ni] = make_frag(pb, pb + 16);
            }
#pragma unroll
            for (int mi = 0; mi < 2; ++mi)
#pragma unroll
                for (int ni = 0; ni < 4; ++ni)
                    acc[mi * 4 + ni] = __builtin_amdgcn_wmma_f32_16x16x32_f16(
                        false, afrag[mi], false, bfrag[ni],
                        (short)0, acc[mi * 4 + ni], false, false);
        }
    }

    // ---- epilogue: stage to LDS, contiguous NCHW stores (+bias) ----
    __syncthreads();
#pragma unroll
    for (int t2 = 0; t2 < 8; ++t2) {
        int mi = t2 >> 2, ni = t2 & 3;
        int col = wave_n * 64 + ni * 16 + l15;
#pragma unroll
        for (int r = 0; r < 8; ++r) {
            int row = wave_m * 32 + mi * 16 + r + (lhi << 3);
            smem.u.C[row * CPAD + col] = acc[t2][r];
        }
    }
    __syncthreads();
#pragma unroll 4
    for (int i = 0; i < 64; ++i) {
        int e   = tid + (i << 8);
        int col = e >> 7;
        int row = e & 127;
        int ocg = nb + col;
        int m   = block_m + row;
        int n   = m >> 12;
        int s   = m & 4095;
        out[(n << 20) + ocg * SPATIAL + s] = smem.u.C[row * CPAD + col] + bias[ocg];
    }
}

// ---------------------------------------------------------------------------
extern "C" void kernel_launch(void* const* d_in, const int* in_sizes, int n_in,
                              void* d_out, int out_size, void* d_ws, size_t ws_size,
                              hipStream_t stream) {
    const float* x      = (const float*)d_in[0];
    const float* conv_w = (const float*)d_in[1];
    const float* conv_b = (const float*)d_in[2];
    const float* lora_A = (const float*)d_in[3];
    const float* lora_B = (const float*)d_in[4];
    float* out = (float*)d_out;

    _Float16* Wt = (_Float16*)d_ws;                              // 1.18 MB
    _Float16* xh = (_Float16*)((char*)d_ws + (2u << 20));        // 33.6 MB

    fuse_weights_kernel<<<(9 * 4 * COUT * 64) / 256, 256, 0, stream>>>(
        conv_w, lora_A, lora_B, Wt);
    convert_x_kernel<<<NIMG * 64 * 4, 256, 0, stream>>>(x, xh);

    dim3 grid(MTOT / BM, COUT / BN);  // 512 x 2
    lora_conv_wmma_kernel<<<grid, 256, 0, stream>>>(xh, Wt, conv_b, out);
}
